// Head_515396076102
// MI455X (gfx1250) — matrix operand order repaired and checked by reading, not compiled, not run on hardware
//
#include <hip/hip_runtime.h>

// MI455X / gfx1250: wave32, WMMA 16x16x32 bf16 (f32 accum).
typedef __attribute__((ext_vector_type(16))) __bf16 v16bf;
typedef __attribute__((ext_vector_type(8)))  float  v8f;

constexpr int Bc = 4;
constexpr int Tc = 4096;
constexpr int Cc = 768;
constexpr int Hc = 64;
constexpr int BT = Bc * Tc;

// C^-0.5 * log2(e): folded into q so attention scores are already in log2
// domain and softmax uses native base-2 v_exp_f32.
#define QSCALE 0.05205877225438f

union Frag16 { v16bf v; __bf16 h[16]; uint4 q[2]; };
union Pack8  { __bf16 h[8]; uint4 q; };

// ---------------------------------------------------------------------------
// Kernel 0: W[C,H] f32 -> Wt[3][H][C] bf16 (transposed so WMMA B-frags are
// contiguous per-lane 32B loads). slice0=Wq, slice1=Wk, slice2=Wv.
// ---------------------------------------------------------------------------
__global__ __launch_bounds__(256) void wt_kernel(const float* __restrict__ Wq,
                                                 const float* __restrict__ Wk,
                                                 const float* __restrict__ Wv,
                                                 __bf16* __restrict__ Wt) {
  int i = blockIdx.x * 256 + threadIdx.x;
  if (i >= 3 * Cc * Hc) return;
  int mat = i / (Cc * Hc);
  int rem = i % (Cc * Hc);
  int r = rem / Hc, col = rem % Hc;
  const float* W = (mat == 0) ? Wq : (mat == 1) ? Wk : Wv;
  Wt[(size_t)mat * Hc * Cc + (size_t)col * Cc + r] = (__bf16)W[rem];
}

// ---------------------------------------------------------------------------
// Kernel 1: QKV projection. blockIdx.y picks {q,k,v}; each wave does a 16-row
// tile: 24 K-steps (C=768) x 4 N-tiles (H=64) of v_wmma_f32_16x16x32_bf16.
// q,k stored row-major bf16 [BT,64]; v stored transposed [B][64][T].
// q is pre-scaled by C^-0.5*log2(e).
// ---------------------------------------------------------------------------
__global__ __launch_bounds__(128) void qkv_proj(const float* __restrict__ x,
                                                const __bf16* __restrict__ Wt,
                                                __bf16* __restrict__ qb,
                                                __bf16* __restrict__ kb,
                                                __bf16* __restrict__ vT) {
  const int w = threadIdx.x >> 5, lane = threadIdx.x & 31;
  const int m = lane & 15, half = lane >> 4;
  const int y = blockIdx.y;
  const int t0 = blockIdx.x * 64 + w * 16;
  const __bf16* Wm = Wt + (size_t)y * Hc * Cc;

  v8f acc[4] = {};
  const float* xr = x + (size_t)(t0 + m) * Cc;  // A: lane m holds row m
  for (int c0 = 0; c0 < Cc; c0 += 32) {
    // 16-bit A layout (16x32): lane run K = half*8 + [0..7] and 16+half*8+[0..7]
    Frag16 a;
    const float* p0 = xr + c0 + half * 8;
    const float* p1 = xr + c0 + 16 + half * 8;
#pragma unroll
    for (int j = 0; j < 8; ++j) {
      a.h[j]     = (__bf16)p0[j];
      a.h[8 + j] = (__bf16)p1[j];
    }
    // Load ALL 4 B-frags first so the 4 WMMAs drain loadcnt incrementally.
    Frag16 bf[4];
#pragma unroll
    for (int j = 0; j < 4; ++j) {  // N-tiles: h0 = j*16, lane n = m
      const __bf16* bp = Wm + (size_t)(j * 16 + m) * Cc + c0 + half * 8;
      bf[j].q[0] = ((const uint4*)bp)[0];
      bf[j].q[1] = ((const uint4*)bp)[2];
    }
#pragma unroll
    for (int j = 0; j < 4; ++j)
      acc[j] = __builtin_amdgcn_wmma_f32_16x16x32_bf16(false, a.v, false, bf[j].v,
                                                       (short)0, acc[j], false, false);
  }

  // C/D layout: acc[j][r] = D[M = r + half*8][N = j*16 + m]
  if (y == 0) {  // q: pre-scale into log2-softmax domain
#pragma unroll
    for (int j = 0; j < 4; ++j) {
      const int col = j * 16 + m;
#pragma unroll
      for (int r = 0; r < 8; ++r)
        qb[(size_t)(t0 + half * 8 + r) * Hc + col] = (__bf16)(acc[j][r] * QSCALE);
    }
  } else if (y == 1) {
#pragma unroll
    for (int j = 0; j < 4; ++j) {
      const int col = j * 16 + m;
#pragma unroll
      for (int r = 0; r < 8; ++r)
        kb[(size_t)(t0 + half * 8 + r) * Hc + col] = (__bf16)acc[j][r];
    }
  } else {
    const int b = t0 / Tc, tl = t0 % Tc;
#pragma unroll
    for (int j = 0; j < 4; ++j) {
      const int h = j * 16 + m;
      Pack8 tmp;
#pragma unroll
      for (int r = 0; r < 8; ++r) tmp.h[r] = (__bf16)acc[j][r];
      // 8 consecutive t values -> one 16B store into vT[b][h][t]
      *(uint4*)(vT + (size_t)(b * Hc + h) * Tc + tl + half * 8) = tmp.q;
    }
  }
}

// ---------------------------------------------------------------------------
// Kernel 2: causal flash attention. One wave owns 16 query rows; key tiles of
// 64 (amortizes the softmax shuffle/exp chain, which co-executing WMMAs make
// the critical path). S = Q.K^T via 8 WMMA, online softmax in log2 domain,
// P routed through per-wave LDS tile (C-layout -> A-layout), O += P.V via
// 8 WMMA against the transposed V buffer.
// ---------------------------------------------------------------------------
__global__ __launch_bounds__(128) void attn(const __bf16* __restrict__ qb,
                                            const __bf16* __restrict__ kb,
                                            const __bf16* __restrict__ vT,
                                            float* __restrict__ out) {
  __shared__ __align__(16) __bf16 pbuf[4][16][72];  // per-wave 16x64 P tile, padded
  const int w = threadIdx.x >> 5, lane = threadIdx.x & 31;
  const int m = lane & 15, half = lane >> 4;
  const int t0g = (blockIdx.x * 4 + w) * 16;
  const int b = t0g / Tc, t0 = t0g % Tc;
  const float NEG_INF = -__builtin_inff();

  // Q A-frags (K=64 -> two 16x32 fragments), resident for the whole loop.
  Frag16 qa[2];
  {
    const __bf16* qr = qb + (size_t)(t0g + m) * Hc;
#pragma unroll
    for (int ks = 0; ks < 2; ++ks) {
      qa[ks].q[0] = *(const uint4*)(qr + ks * 32 + half * 8);
      qa[ks].q[1] = *(const uint4*)(qr + ks * 32 + 16 + half * 8);
    }
  }

  float rmax[8], rsum[8];
#pragma unroll
  for (int r = 0; r < 8; ++r) { rmax[r] = NEG_INF; rsum[r] = 0.f; }
  v8f o[4] = {};

  const __bf16* krow = kb + (size_t)b * Tc * Hc;
  const __bf16* vrow = vT + (size_t)b * Hc * Tc;

  for (int s0 = 0; s0 <= t0 + 15; s0 += 64) {   // 64-key tiles; T%64==0 -> no OOB
    if (s0 + 64 + m + 95 < Tc)  // prefetch next key tile (global_prefetch path)
      __builtin_prefetch(krow + (size_t)(s0 + 64 + m) * Hc, 0, 1);

    // ---- S = Q.K^T : load all 8 K B-frags, then 8 WMMAs ----
    Frag16 kf[4][2];  // [n-subtile][k-step]; B(h,n)=k[n][h]: contiguous per lane
#pragma unroll
    for (int j = 0; j < 4; ++j) {
      const __bf16* kr = krow + (size_t)(s0 + j * 16 + m) * Hc;
      kf[j][0].q[0] = *(const uint4*)(kr + half * 8);
      kf[j][0].q[1] = *(const uint4*)(kr + half * 8 + 16);
      kf[j][1].q[0] = *(const uint4*)(kr + 32 + half * 8);
      kf[j][1].q[1] = *(const uint4*)(kr + 32 + half * 8 + 16);
    }
    v8f sacc[4] = {};
#pragma unroll
    for (int j = 0; j < 4; ++j) {
      sacc[j] = __builtin_amdgcn_wmma_f32_16x16x32_bf16(false, qa[0].v, false, kf[j][0].v,
                                                        (short)0, sacc[j], false, false);
      sacc[j] = __builtin_amdgcn_wmma_f32_16x16x32_bf16(false, qa[1].v, false, kf[j][1].v,
                                                        (short)0, sacc[j], false, false);
    }

    // ---- causal mask (scores already scaled into log2 domain via q) ----
#pragma unroll
    for (int j = 0; j < 4; ++j)
#pragma unroll
      for (int r = 0; r < 8; ++r) {
        const int ts = s0 + j * 16 + m;
        const int tq = t0 + half * 8 + r;
        sacc[j][r] = (ts <= tq) ? sacc[j][r] : NEG_INF;
      }

    // ---- online softmax, base-2 (rows live across 16-lane halves) ----
    float corr[8];
#pragma unroll
    for (int r = 0; r < 8; ++r) {
      float lm = fmaxf(fmaxf(sacc[0][r], sacc[1][r]), fmaxf(sacc[2][r], sacc[3][r]));
#pragma unroll
      for (int d = 1; d < 16; d <<= 1) lm = fmaxf(lm, __shfl_xor(lm, d, 32));
      const float nm = fmaxf(rmax[r], lm);
      corr[r] = __builtin_exp2f(rmax[r] - nm);
      rmax[r] = nm;
      float ls = 0.f;
#pragma unroll
      for (int j = 0; j < 4; ++j) {
        const float p = __builtin_exp2f(sacc[j][r] - nm);
        sacc[j][r] = p;
        ls += p;
      }
#pragma unroll
      for (int d = 1; d < 16; d <<= 1) ls += __shfl_xor(ls, d, 32);
      rsum[r] = rsum[r] * corr[r] + ls;
    }
#pragma unroll
    for (int j = 0; j < 4; ++j)
#pragma unroll
      for (int r = 0; r < 8; ++r) o[j][r] *= corr[r];

    // ---- P (16x64): C-layout -> A-layout via per-wave LDS tile ----
#pragma unroll
    for (int j = 0; j < 4; ++j)
#pragma unroll
      for (int r = 0; r < 8; ++r)
        pbuf[w][half * 8 + r][j * 16 + m] = (__bf16)sacc[j][r];
    asm volatile("" ::: "memory");  // same-wave DS ops are in-order in HW
    Frag16 pa[2];
#pragma unroll
    for (int ks = 0; ks < 2; ++ks) {
      pa[ks].q[0] = *(const uint4*)&pbuf[w][m][ks * 32 + half * 8];
      pa[ks].q[1] = *(const uint4*)&pbuf[w][m][ks * 32 + 16 + half * 8];
    }
    asm volatile("" ::: "memory");

    // ---- O += P.V : load all 8 V B-frags, then 8 WMMAs ----
    Frag16 vf[2][4];  // [k-step][h-tile]
#pragma unroll
    for (int ks = 0; ks < 2; ++ks) {
      const int sv = s0 + ks * 32 + half * 8;
#pragma unroll
      for (int j = 0; j < 4; ++j) {
        const __bf16* vp = vrow + (size_t)(j * 16 + m) * Tc + sv;
        vf[ks][j].q[0] = ((const uint4*)vp)[0];
        vf[ks][j].q[1] = ((const uint4*)vp)[2];
      }
    }
#pragma unroll
    for (int j = 0; j < 4; ++j) {
      o[j] = __builtin_amdgcn_wmma_f32_16x16x32_bf16(false, pa[0].v, false, vf[0][j].v,
                                                     (short)0, o[j], false, false);
      o[j] = __builtin_amdgcn_wmma_f32_16x16x32_bf16(false, pa[1].v, false, vf[1][j].v,
                                                     (short)0, o[j], false, false);
    }
  }

  // ---- epilogue: normalize (8 reciprocals, not 32 divides) and store f32 ----
  float inv[8];
#pragma unroll
  for (int r = 0; r < 8; ++r) inv[r] = 1.0f / rsum[r];
#pragma unroll
  for (int j = 0; j < 4; ++j)
#pragma unroll
    for (int r = 0; r < 8; ++r)
      out[(size_t)(t0g + half * 8 + r) * Hc + j * 16 + m] = o[j][r] * inv[r];
}

// ---------------------------------------------------------------------------
extern "C" void kernel_launch(void* const* d_in, const int* in_sizes, int n_in,
                              void* d_out, int out_size, void* d_ws, size_t ws_size,
                              hipStream_t stream) {
  (void)in_sizes; (void)n_in; (void)out_size; (void)ws_size;
  const float* x  = (const float*)d_in[0];
  const float* Wk = (const float*)d_in[1];
  const float* Wq = (const float*)d_in[2];
  const float* Wv = (const float*)d_in[3];
  float* out = (float*)d_out;

  __bf16* ws = (__bf16*)d_ws;
  __bf16* qb = ws;                        // [BT,64] bf16 (pre-scaled)
  __bf16* kb = qb + (size_t)BT * Hc;      // [BT,64] bf16
  __bf16* vT = kb + (size_t)BT * Hc;      // [B,64,T] bf16
  __bf16* Wt = vT + (size_t)BT * Hc;      // [3,64,768] bf16

  const int nW = 3 * Cc * Hc;
  wt_kernel<<<(nW + 255) / 256, 256, 0, stream>>>(Wq, Wk, Wv, Wt);
  qkv_proj<<<dim3(BT / 64, 3), 128, 0, stream>>>(x, Wt, qb, kb, vT);
  attn<<<BT / 64, 128, 0, stream>>>(qb, kb, vT, out);
}
